// XPaiNN_41893111005214
// MI455X (gfx1250) — matrix operand order repaired and checked by reading, not compiled, not run on hardware
//
#include <hip/hip_runtime.h>

// ---------------- problem constants ----------------
constexpr int kN  = 10000;
constexpr int kE  = 160000;
constexpr int kF  = 128;
constexpr int kNB = 20;
constexpr int kBlocks = 3;
constexpr float kCutoff = 5.0f;
constexpr float kEps = 1e-8f;
constexpr float kPi = 3.14159265358979323846f;

// ---------------- WMMA types ----------------
typedef __attribute__((ext_vector_type(16))) __bf16 v16bf;
typedef __attribute__((ext_vector_type(8)))  __bf16 v8bf;
typedef __attribute__((ext_vector_type(8)))  float  v8f;

static __device__ __forceinline__ float silu(float x) {
    return x / (1.0f + __expf(-x));
}

// ---------------------------------------------------------------------------
// Weight pre-pack: f32 [K x Nld] row-major -> bf16 in WMMA B-fragment lane
// order: Bp[(kt*(Nld/16)+nt)*32 + lane][0..15], where lane covers column
// nt*16+(lane&15) and K rows kt*32 + (lane>>4)*16 + i.  One thread per
// (kt,nt,lane) group writes 16 contiguous bf16 (32 bytes).
// ---------------------------------------------------------------------------
__global__ void pack_b_kernel(const float* __restrict__ W,
                              __bf16* __restrict__ Bp, int K, int Nld)
{
    int g = blockIdx.x * blockDim.x + threadIdx.x;
    int total = (K / 32) * (Nld / 16) * 32;
    if (g >= total) return;
    int lane = g & 31;
    int t = g >> 5;
    int nt = t % (Nld / 16);
    int kt = t / (Nld / 16);
    int kbase = kt * 32 + (lane >> 4) * 16;
    int ncol  = nt * 16 + (lane & 15);
    __bf16* out = Bp + (size_t)g * 16;
#pragma unroll
    for (int i = 0; i < 16; ++i)
        out[i] = (__bf16)W[(size_t)(kbase + i) * Nld + ncol];
}

// ---------------------------------------------------------------------------
// elementwise f32 -> bf16 (count multiple of 4)
// ---------------------------------------------------------------------------
__global__ void cvt_bf16_kernel(const float* __restrict__ in,
                                __bf16* __restrict__ out, int count4)
{
    int i = blockIdx.x * blockDim.x + threadIdx.x;
    if (i >= count4) return;
    int b = i * 4;
#pragma unroll
    for (int j = 0; j < 4; ++j) out[b + j] = (__bf16)in[b + j];
}

// ---------------------------------------------------------------------------
// Generic GEMM:  Out[M x Nld] = A_bf16[M x K] * Bpacked[K x Nld] (+ bias)
// One wave -> one 16x16 tile; K templated so the loop fully unrolls:
// per K-step = 2x b128 A loads + 2x b128 packed-B loads + 1 wmma.
// Grid (M/16, Nld/128), 256 threads (8 waves x 16 columns each).
// EPI_SILU_BF16: epilogue writes silu(out) as bf16 (feeds next GEMM),
// otherwise f32 (+bias).
// ---------------------------------------------------------------------------
template <int K, bool EPI_SILU_BF16>
__global__ __launch_bounds__(256) void gemm_wmma(
    const __bf16* __restrict__ A, const __bf16* __restrict__ Bp,
    const float* __restrict__ bias, float* __restrict__ OutF,
    __bf16* __restrict__ OutB, int Nld)
{
    const int lane  = threadIdx.x & 31;
    const int wave  = threadIdx.x >> 5;
    const int row0  = blockIdx.x * 16;
    const int col0  = blockIdx.y * 128 + wave * 16;
    const int lmod  = lane & 15;
    const int khalf = lane >> 4;
    const int m  = row0 + lmod;
    const int n  = col0 + lmod;
    const int NT = Nld / 16;
    const int ct = col0 >> 4;

    v8f acc = {};
#pragma unroll
    for (int kt = 0; kt < K / 32; ++kt) {
        // A fragment: lanes 0-15 hold K kt*32+[0..15];
        //             lanes 16-31 hold +[16..31]
        const __bf16* arow = A + (size_t)m * K + kt * 32 + khalf * 16;
        v8bf a0 = *(const v8bf*)arow;
        v8bf a1 = *(const v8bf*)(arow + 8);
        v16bf afrag = __builtin_shufflevector(a0, a1,
            0, 1, 2, 3, 4, 5, 6, 7, 8, 9, 10, 11, 12, 13, 14, 15);

        // B fragment: pre-packed contiguous 32B per lane
        const __bf16* bbase = Bp + (((size_t)kt * NT + ct) * 32 + lane) * 16;
        v8bf b0 = *(const v8bf*)bbase;
        v8bf b1 = *(const v8bf*)(bbase + 8);
        v16bf bfrag = __builtin_shufflevector(b0, b1,
            0, 1, 2, 3, 4, 5, 6, 7, 8, 9, 10, 11, 12, 13, 14, 15);

        acc = __builtin_amdgcn_wmma_f32_16x16x32_bf16(
            false, afrag, false, bfrag, (short)0, acc, false, false);
    }

    const float bv = bias ? bias[n] : 0.0f;
    const int mo = row0 + 8 * khalf;   // C layout: elem r -> row r + 8*(lane>>4)
#pragma unroll
    for (int r = 0; r < 8; ++r) {
        float val = acc[r] + bv;
        if (EPI_SILU_BF16)
            OutB[(size_t)(mo + r) * Nld + n] = (__bf16)silu(val);
        else
            OutF[(size_t)(mo + r) * Nld + n] = val;
    }
}

// ---------------------------------------------------------------------------
// x init:  x = z_embed[z]
// ---------------------------------------------------------------------------
__global__ void init_x_kernel(const int* __restrict__ z,
                              const float* __restrict__ z_embed,
                              float* __restrict__ x)
{
    int idx = blockIdx.x * blockDim.x + threadIdx.x;
    if (idx >= kN * kF) return;
    int n = idx / kF, f = idx % kF;
    x[idx] = z_embed[(size_t)z[n] * kF + f];
}

// ---------------------------------------------------------------------------
// Edge geometry: d, rhat, rbf, fc
// ---------------------------------------------------------------------------
__global__ void edge_geom_kernel(const float* __restrict__ pos,
                                 const int* __restrict__ ei,
                                 float* __restrict__ rhat,
                                 float* __restrict__ rbf,
                                 float* __restrict__ fc)
{
    int e = blockIdx.x * blockDim.x + threadIdx.x;
    if (e >= kE) return;
    int s = ei[e], dt = ei[kE + e];
    float rv[3]; float d2 = kEps;
#pragma unroll
    for (int c = 0; c < 3; ++c) {
        rv[c] = pos[dt * 3 + c] - pos[s * 3 + c];
        d2 += rv[c] * rv[c];
    }
    float d = sqrtf(d2);
    float inv = 1.0f / d;
#pragma unroll
    for (int c = 0; c < 3; ++c) rhat[e * 3 + c] = rv[c] * inv;
    float arg = kPi * d / kCutoff;
#pragma unroll
    for (int j = 0; j < kNB; ++j)
        rbf[(size_t)e * kNB + j] = __sinf((float)(j + 1) * arg) * inv;
    fc[e] = (d < kCutoff) ? 0.5f * (__cosf(arg) + 1.0f) : 0.0f;
}

// ---------------------------------------------------------------------------
// LayerNorm: one wave32 per node, 4 elems/lane; writes bf16 (GEMM A input)
// ---------------------------------------------------------------------------
__global__ __launch_bounds__(256) void layernorm_kernel(
    const float* __restrict__ x, const float* __restrict__ g,
    const float* __restrict__ b, __bf16* __restrict__ out)
{
    int wave = threadIdx.x >> 5, lane = threadIdx.x & 31;
    int node = blockIdx.x * 8 + wave;
    if (node >= kN) return;
    const float* row = x + (size_t)node * kF;
    float vals[4]; float s = 0.0f;
#pragma unroll
    for (int k = 0; k < 4; ++k) { vals[k] = row[lane + 32 * k]; s += vals[k]; }
#pragma unroll
    for (int m = 16; m >= 1; m >>= 1) s += __shfl_xor(s, m, 32);
    float mu = s * (1.0f / kF);
    float q = 0.0f;
#pragma unroll
    for (int k = 0; k < 4; ++k) { float d = vals[k] - mu; q += d * d; }
#pragma unroll
    for (int m = 16; m >= 1; m >>= 1) q += __shfl_xor(q, m, 32);
    float inv = rsqrtf(q * (1.0f / kF) + 1e-5f);
    __bf16* orow = out + (size_t)node * kF;
#pragma unroll
    for (int k = 0; k < 4; ++k) {
        int f = lane + 32 * k;
        orow[f] = (__bf16)((vals[k] - mu) * inv * g[f] + b[f]);
    }
}

// ---------------------------------------------------------------------------
// Edge messages: Wf = (rbf@Wrbf + brbf)*fc ; m = phi[src]*Wf ; scatter to dst
// ---------------------------------------------------------------------------
__global__ __launch_bounds__(128) void edge_msg_kernel(
    const int* __restrict__ ei,
    const float* __restrict__ phi, const float* __restrict__ rbf,
    const float* __restrict__ fc,  const float* __restrict__ rhat,
    const float* __restrict__ v,
    const float* __restrict__ Wrbf, const float* __restrict__ brbf,
    float* __restrict__ dx, float* __restrict__ dv)
{
    int e = blockIdx.x;
    int f = threadIdx.x;
    int s = ei[e], dt = ei[kE + e];
    float w0 = brbf[f], w1 = brbf[kF + f], w2 = brbf[2 * kF + f];
    const float* rb = rbf + (size_t)e * kNB;
#pragma unroll
    for (int j = 0; j < kNB; ++j) {
        float r = rb[j];
        const float* wr = Wrbf + (size_t)j * 3 * kF;
        w0 += r * wr[f];
        w1 += r * wr[kF + f];
        w2 += r * wr[2 * kF + f];
    }
    float fce = fc[e];
    w0 *= fce; w1 *= fce; w2 *= fce;
    const float* ph = phi + (size_t)s * 3 * kF;
    float ms  = ph[f] * w0;
    float mvv = ph[kF + f] * w1;
    float mvs = ph[2 * kF + f] * w2;
    atomicAdd(&dx[(size_t)dt * kF + f], ms);
#pragma unroll
    for (int c = 0; c < 3; ++c) {
        float ev = mvv * v[((size_t)s * 3 + c) * kF + f] + mvs * rhat[e * 3 + c];
        atomicAdd(&dv[((size_t)dt * 3 + c) * kF + f], ev);
    }
}

// ---------------------------------------------------------------------------
__global__ void add_inplace_kernel(float* __restrict__ dst,
                                   const float* __restrict__ src, int count)
{
    int i = blockIdx.x * blockDim.x + threadIdx.x;
    if (i < count) dst[i] += src[i];
}

// vnorm / dot(Uv,Vv) / build bf16 concat[xn2, vnorm]
__global__ void catdot_kernel(const float* __restrict__ Uv,
                              const float* __restrict__ Vv,
                              const __bf16* __restrict__ xn,
                              __bf16* __restrict__ cat,
                              float* __restrict__ dot)
{
    int idx = blockIdx.x * blockDim.x + threadIdx.x;
    if (idx >= kN * kF) return;
    int n = idx / kF, f = idx % kF;
    float u0 = Uv[((size_t)n * 3 + 0) * kF + f];
    float u1 = Uv[((size_t)n * 3 + 1) * kF + f];
    float u2 = Uv[((size_t)n * 3 + 2) * kF + f];
    float w0 = Vv[((size_t)n * 3 + 0) * kF + f];
    float w1 = Vv[((size_t)n * 3 + 1) * kF + f];
    float w2 = Vv[((size_t)n * 3 + 2) * kF + f];
    dot[idx] = u0 * w0 + u1 * w1 + u2 * w2;
    cat[(size_t)n * 2 * kF + kF + f] = (__bf16)sqrtf(w0 * w0 + w1 * w1 + w2 * w2 + kEps);
    cat[(size_t)n * 2 * kF + f] = xn[idx];
}

// x += asv*dot + ass ; v += avv*Uv
__global__ void update_kernel(float* __restrict__ x, float* __restrict__ v,
                              const float* __restrict__ a,
                              const float* __restrict__ dot,
                              const float* __restrict__ Uv)
{
    int idx = blockIdx.x * blockDim.x + threadIdx.x;
    if (idx >= kN * kF) return;
    int n = idx / kF, f = idx % kF;
    const float* ar = a + (size_t)n * 3 * kF;
    float avv = ar[f], asv = ar[kF + f], ass = ar[2 * kF + f];
    x[idx] += asv * dot[idx] + ass;
#pragma unroll
    for (int c = 0; c < 3; ++c)
        v[((size_t)n * 3 + c) * kF + f] += avv * Uv[((size_t)n * 3 + c) * kF + f];
}

// e_atom = silu(t1) @ Wo2 + bo2 ; global sum
__global__ __launch_bounds__(128) void out_reduce_kernel(
    const float* __restrict__ t1, const float* __restrict__ Wo2,
    const float* __restrict__ bo2, float* __restrict__ out)
{
    __shared__ float red[128];
    int n = blockIdx.x, f = threadIdx.x;
    float t = t1[(size_t)n * kF + f];
    red[f] = silu(t) * Wo2[f];
    __syncthreads();
#pragma unroll
    for (int off = 64; off >= 1; off >>= 1) {
        if (f < off) red[f] += red[f + off];
        __syncthreads();
    }
    if (f == 0) atomicAdd(out, red[0] + bo2[0]);
}

// ---------------------------------------------------------------------------
extern "C" void kernel_launch(void* const* d_in, const int* in_sizes, int n_in,
                              void* d_out, int out_size, void* d_ws, size_t ws_size,
                              hipStream_t stream)
{
    const float* pos     = (const float*)d_in[0];
    const int*   z       = (const int*)  d_in[1];
    const int*   ei      = (const int*)  d_in[2];
    const float* z_embed = (const float*)d_in[3];
    const float* Wrbf    = (const float*)d_in[4];
    const float* brbf    = (const float*)d_in[5];
    const float* ln_g    = (const float*)d_in[6];
    const float* ln_b    = (const float*)d_in[7];
    const float* Wm1     = (const float*)d_in[8];
    const float* bm1     = (const float*)d_in[9];
    const float* Wm2     = (const float*)d_in[10];
    const float* bm2     = (const float*)d_in[11];
    const float* Uw      = (const float*)d_in[12];
    const float* Vw      = (const float*)d_in[13];
    const float* ln2_g   = (const float*)d_in[14];
    const float* ln2_b   = (const float*)d_in[15];
    const float* Wu1     = (const float*)d_in[16];
    const float* bu1     = (const float*)d_in[17];
    const float* Wu2     = (const float*)d_in[18];
    const float* bu2     = (const float*)d_in[19];
    const float* Wo1     = (const float*)d_in[20];
    const float* bo1     = (const float*)d_in[21];
    const float* Wo2     = (const float*)d_in[22];
    const float* bo2     = (const float*)d_in[23];

    const size_t NF = (size_t)kN * kF;
    const size_t FF = (size_t)kF * kF;

    // ---- f32 workspace ----
    float* ws   = (float*)d_ws;
    float* x    = ws;                 // N*F
    float* v    = x    + NF;          // 3N*F
    float* phi  = v    + 3 * NF;      // N*3F   (reused as t1 at readout)
    float* dx   = phi  + 3 * NF;      // N*F
    float* dv   = dx   + NF;          // 3N*F
    float* Uv   = dv   + 3 * NF;      // 3N*F
    float* Vv   = Uv   + 3 * NF;      // 3N*F
    float* dot  = Vv   + 3 * NF;      // N*F
    float* aM   = dot  + NF;          // N*3F
    float* rhat = aM   + 3 * NF;      // 3E
    float* rbf  = rhat + 3 * (size_t)kE;   // E*NB
    float* fcb  = rbf  + (size_t)kNB * kE; // E
    float* fend = fcb  + kE;

    // ---- bf16 workspace (16B-aligned: f32 count above is a multiple of 4) ----
    __bf16* bws    = (__bf16*)fend;
    __bf16* xn_bf  = bws;              // N*F
    __bf16* h_bf   = xn_bf  + NF;      // N*F
    __bf16* cat_bf = h_bf   + NF;      // N*2F
    __bf16* v_bf   = cat_bf + 2 * NF;  // 3N*F
    __bf16* x_bf   = v_bf   + 3 * NF;  // N*F
    __bf16* pWm1   = x_bf   + NF;            // 3*F*F
    __bf16* pWm2   = pWm1   + 3 * FF;        // 3*F*3F
    __bf16* pUw    = pWm2   + 9 * FF;        // 3*F*F
    __bf16* pVw    = pUw    + 3 * FF;        // 3*F*F
    __bf16* pWu1   = pVw    + 3 * FF;        // 3*2F*F
    __bf16* pWu2   = pWu1   + 6 * FF;        // 3*F*3F
    __bf16* pWo1   = pWu2   + 9 * FF;        // F*F

    hipMemsetAsync(d_out, 0, sizeof(float), stream);
    hipMemsetAsync(v, 0, 3 * NF * sizeof(float), stream);

    // ---- pre-pack all weights to WMMA-fragment bf16 layout ----
    auto packB = [&](const float* W, __bf16* P, int K, int Nld) {
        int total = (K / 32) * (Nld / 16) * 32;
        pack_b_kernel<<<(total + 255) / 256, 256, 0, stream>>>(W, P, K, Nld);
    };
    for (int b = 0; b < kBlocks; ++b) {
        packB(Wm1 + b * FF,         pWm1 + b * FF,         kF,     kF);
        packB(Wm2 + b * 3 * FF,     pWm2 + b * 3 * FF,     kF,     3 * kF);
        packB(Uw  + b * FF,         pUw  + b * FF,         kF,     kF);
        packB(Vw  + b * FF,         pVw  + b * FF,         kF,     kF);
        packB(Wu1 + b * 2 * FF,     pWu1 + b * 2 * FF,     2 * kF, kF);
        packB(Wu2 + b * 3 * FF,     pWu2 + b * 3 * FF,     kF,     3 * kF);
    }
    packB(Wo1, pWo1, kF, kF);

    init_x_kernel<<<(kN * kF + 255) / 256, 256, 0, stream>>>(z, z_embed, x);
    edge_geom_kernel<<<(kE + 255) / 256, 256, 0, stream>>>(pos, ei, rhat, rbf, fcb);

    const dim3 blk(256);
    const int cvtBlocks3NF = (int)((3 * NF / 4 + 255) / 256);
    const int cvtBlocksNF  = (int)((NF / 4 + 255) / 256);

    for (int b = 0; b < kBlocks; ++b) {
        // phi = silu(LN(x) @ Wm1 + bm1) @ Wm2 + bm2
        layernorm_kernel<<<kN / 8, blk, 0, stream>>>(x, ln_g + b * kF, ln_b + b * kF, xn_bf);
        gemm_wmma<kF, true><<<dim3(kN / 16, 1), blk, 0, stream>>>(
            xn_bf, pWm1 + b * FF, bm1 + b * kF, nullptr, h_bf, kF);
        gemm_wmma<kF, false><<<dim3(kN / 16, 3), blk, 0, stream>>>(
            h_bf, pWm2 + b * 3 * FF, bm2 + b * 3 * kF, phi, nullptr, 3 * kF);

        // edge messages -> dx, dv
        hipMemsetAsync(dx, 0, NF * sizeof(float), stream);
        hipMemsetAsync(dv, 0, 3 * NF * sizeof(float), stream);
        edge_msg_kernel<<<kE, 128, 0, stream>>>(
            ei, phi, rbf, fcb, rhat, v,
            Wrbf + (size_t)b * kNB * 3 * kF, brbf + b * 3 * kF, dx, dv);
        add_inplace_kernel<<<(int)((NF + 255) / 256), blk, 0, stream>>>(x, dx, (int)NF);
        add_inplace_kernel<<<(int)((3 * NF + 255) / 256), blk, 0, stream>>>(v, dv, (int)(3 * NF));

        // Uv = v @ Uw ; Vv = v @ Vw   (rows = 3N, bf16 copy of v)
        cvt_bf16_kernel<<<cvtBlocks3NF, blk, 0, stream>>>(v, v_bf, (int)(3 * NF / 4));
        gemm_wmma<kF, false><<<dim3(3 * kN / 16, 1), blk, 0, stream>>>(
            v_bf, pUw + b * FF, nullptr, Uv, nullptr, kF);
        gemm_wmma<kF, false><<<dim3(3 * kN / 16, 1), blk, 0, stream>>>(
            v_bf, pVw + b * FF, nullptr, Vv, nullptr, kF);

        // a = silu([LN2(x), |Vv|] @ Wu1 + bu1) @ Wu2 + bu2
        layernorm_kernel<<<kN / 8, blk, 0, stream>>>(x, ln2_g + b * kF, ln2_b + b * kF, xn_bf);
        catdot_kernel<<<(kN * kF + 255) / 256, blk, 0, stream>>>(Uv, Vv, xn_bf, cat_bf, dot);
        gemm_wmma<2 * kF, true><<<dim3(kN / 16, 1), blk, 0, stream>>>(
            cat_bf, pWu1 + b * 2 * FF, bu1 + b * kF, nullptr, h_bf, kF);
        gemm_wmma<kF, false><<<dim3(kN / 16, 3), blk, 0, stream>>>(
            h_bf, pWu2 + b * 3 * FF, bu2 + b * 3 * kF, aM, nullptr, 3 * kF);

        update_kernel<<<(kN * kF + 255) / 256, blk, 0, stream>>>(x, v, aM, dot, Uv);
    }

    // readout: sum( silu(x@Wo1+bo1) @ Wo2 + bo2 )
    cvt_bf16_kernel<<<cvtBlocksNF, blk, 0, stream>>>(x, x_bf, (int)(NF / 4));
    gemm_wmma<kF, false><<<dim3(kN / 16, 1), blk, 0, stream>>>(
        x_bf, pWo1, bo1, phi /*t1*/, nullptr, kF);
    out_reduce_kernel<<<kN, 128, 0, stream>>>(phi, Wo2, bo2, (float*)d_out);
}
